// GTLayer_1357209665642
// MI455X (gfx1250) — matrix operand order repaired and checked by reading, not compiled, not run on hardware
//
#include <hip/hip_runtime.h>

#define N_NODES 40000
#define N_EDGES 640000
#define HIDDEN 128
#define HEADS 8
#define SCALING 0.25f          /* 16^-0.5 */
#define NBIG ((size_t)N_NODES * HIDDEN)   /* 5,120,000 == N_EDGES*HEADS */

typedef __attribute__((ext_vector_type(16))) __bf16 v16bf;
typedef __attribute__((ext_vector_type(8)))  __bf16 v8bf;
typedef __attribute__((ext_vector_type(8)))  float  v8f;

static __device__ inline __bf16 bf(float f) { return (__bf16)f; }

static __device__ inline v8f wmma_bf16(v16bf a, v16bf b, v8f c) {
    return __builtin_amdgcn_wmma_f32_16x16x32_bf16(false, a, false, b, (short)0, c, false, false);
}

// A-matrix (16x32 bf16) tile loader from row-major f32 [ld]: lane&15 = M,
// lanes>=16 shift K by +8; elements 0..7 -> K base..base+7, 8..15 -> K base+16..base+23
static __device__ inline v16bf load_a_f32(const float* base, int ld, int row0, int k0, int lane) {
    int m  = lane & 15;
    int kb = k0 + ((lane >> 4) << 3);
    const float* p = base + (size_t)(row0 + m) * ld + kb;
    v16bf a;
#pragma unroll
    for (int e = 0; e < 8; ++e) a[e] = bf(p[e]);
#pragma unroll
    for (int e = 0; e < 8; ++e) a[8 + e] = bf(p[16 + e]);
    return a;
}

// B-matrix (32x16 bf16) tile from pre-converted bf16 weight: lane&15 = N,
// lanes>=16 shift K by +8; elements 0..7 -> K base..base+7, 8..15 -> K base+16..base+23
static __device__ inline v16bf load_b_bf(const __bf16* W, int ld, int k0, int n0, int lane) {
    int n  = n0 + (lane & 15);
    int kb = k0 + ((lane >> 4) << 3);
    v16bf b;
#pragma unroll
    for (int e = 0; e < 8; ++e) b[e] = W[(size_t)(kb + e) * ld + n];
#pragma unroll
    for (int e = 0; e < 8; ++e) b[8 + e] = W[(size_t)(kb + 16 + e) * ld + n];
    return b;
}

// 16x128 @ 128x128 tile GEMM: out[row0..row0+15, 0..127] = (A@W + bias)*scale (+ resid)
static __device__ inline void gemm16_128(const v16bf a[4], const __bf16* W, const float* bias,
                                         float scale, const float* resid, float* out,
                                         int row0, int lane) {
    int nlo = lane & 15, mhi = (lane >> 4) << 3;
#pragma unroll
    for (int nt = 0; nt < 8; ++nt) {
        v8f acc = {};
#pragma unroll
        for (int kt = 0; kt < 4; ++kt) {
            v16bf b = load_b_bf(W, 128, kt * 32, nt * 16, lane);
            acc = wmma_bf16(a[kt], b, acc);
        }
        int ncol = nt * 16 + nlo;
        float bs = bias[ncol];
#pragma unroll
        for (int r = 0; r < 8; ++r) {
            size_t idx = (size_t)(row0 + r + mhi) * 128 + ncol;
            float v = (acc[r] + bs) * scale;
            if (resid) v += resid[idx];
            out[idx] = v;
        }
    }
}

// ---------------- kernels ----------------

__global__ void fill_kernel(float* p, long n, float v) {
    long i = (long)blockIdx.x * blockDim.x + threadIdx.x;
    long stride = (long)gridDim.x * blockDim.x;
    for (; i < n; i += stride) p[i] = v;
}

__global__ void cvt_bf16_kernel(const float* __restrict__ src, __bf16* __restrict__ dst, int n) {
    int i = blockIdx.x * blockDim.x + threadIdx.x;
    int stride = gridDim.x * blockDim.x;
    for (; i < n; i += stride) dst[i] = bf(src[i]);
}

__global__ void __launch_bounds__(128) qkv_kernel(const float* __restrict__ h,
        const __bf16* Wq, const float* bq, const __bf16* Wk, const float* bk,
        const __bf16* Wv, const float* bv, float* Q, float* K, float* V) {
    int wave = threadIdx.x >> 5, lane = threadIdx.x & 31;
    int tile = blockIdx.x * 4 + wave;           // 625*4 = 2500 tiles exactly
    int row0 = tile * 16;
    v16bf a[4];
#pragma unroll
    for (int kt = 0; kt < 4; ++kt) a[kt] = load_a_f32(h, 128, row0, kt * 32, lane);
    gemm16_128(a, Wq, bq, SCALING, nullptr, Q, row0, lane);
    gemm16_128(a, Wk, bk, 1.0f,    nullptr, K, row0, lane);
    gemm16_128(a, Wv, bv, 1.0f,    nullptr, V, row0, lane);
}

__global__ void __launch_bounds__(128) proj_kernel(const float* __restrict__ AO,
        const __bf16* Wo, const float* bo, const float* __restrict__ h, float* T) {
    int wave = threadIdx.x >> 5, lane = threadIdx.x & 31;
    int tile = blockIdx.x * 4 + wave;
    int row0 = tile * 16;
    v16bf a[4];
#pragma unroll
    for (int kt = 0; kt < 4; ++kt) a[kt] = load_a_f32(AO, 128, row0, kt * 32, lane);
    gemm16_128(a, Wo, bo, 1.0f, h, T, row0, lane);
}

static __device__ inline void atomicMaxFloat(float* addr, float val) {
    if (val >= 0.0f) atomicMax((int*)addr, __float_as_int(val));
    else             atomicMin((unsigned int*)addr, __float_as_uint(val));
}

// wave per edge: s[e,h] = sum_d q[row,d,h]*k[col,d,h]; flat idx = d*8+h
__global__ void __launch_bounds__(256) edge_score_kernel(const float* __restrict__ Q,
        const float* __restrict__ K, const int* __restrict__ row, const int* __restrict__ col,
        float* S, float* Mx) {
    int wid  = (int)((blockIdx.x * (size_t)blockDim.x + threadIdx.x) >> 5);
    if (wid >= N_EDGES) return;
    int lane = threadIdx.x & 31;
    int r = row[wid], c = col[wid];
    int hh = lane & 7, qd = lane >> 3;
    const float* qp = Q + (size_t)r * 128 + qd * 32 + hh;
    const float* kp = K + (size_t)c * 128 + qd * 32 + hh;
    float p = 0.f;
#pragma unroll
    for (int j = 0; j < 4; ++j) p += qp[j * 8] * kp[j * 8];
    p += __shfl_xor(p, 8, 32);
    p += __shfl_xor(p, 16, 32);
    if (lane < 8) {
        S[(size_t)wid * 8 + hh] = p;
        atomicMaxFloat(&Mx[(size_t)r * 8 + hh], p);
    }
}

__global__ void edge_exp_kernel(float* S, const float* __restrict__ Mx, float* Z,
                                const int* __restrict__ row) {
    long n = (long)N_EDGES * 8;
    long i = (long)blockIdx.x * blockDim.x + threadIdx.x;
    long stride = (long)gridDim.x * blockDim.x;
    for (; i < n; i += stride) {
        int e = (int)(i >> 3), hh = (int)(i & 7);
        int r = row[e];
        float v = expf(S[i] - Mx[(size_t)r * 8 + hh]);
        S[i] = v;
        atomicAdd(&Z[(size_t)r * 8 + hh], v);
    }
}

// wave per edge: AO[row, f] += (S[e,h]/Z[row,h]) * V[col, f], f = 4*lane + j, h = f&7
__global__ void __launch_bounds__(256) edge_spmm_kernel(const float* __restrict__ S,
        const float* __restrict__ Z, const float* __restrict__ V,
        const int* __restrict__ row, const int* __restrict__ col, float* AO) {
    int wid = (int)((blockIdx.x * (size_t)blockDim.x + threadIdx.x) >> 5);
    if (wid >= N_EDGES) return;
    int lane = threadIdx.x & 31;
    int r = row[wid], c = col[wid];
    int f0 = lane * 4;
#pragma unroll
    for (int j = 0; j < 4; ++j) {
        int f = f0 + j, hh = f & 7;
        float attn = S[(size_t)wid * 8 + hh] / Z[(size_t)r * 8 + hh];
        atomicAdd(&AO[(size_t)r * 128 + f], attn * V[(size_t)c * 128 + f]);
    }
}

__global__ void __launch_bounds__(128) bn_stats_kernel(const float* __restrict__ X,
                                                       float* sum, float* sq) {
    int c = threadIdx.x;   // 128 channels
    float s = 0.f, s2 = 0.f;
    for (int rrow = blockIdx.x; rrow < N_NODES; rrow += gridDim.x) {
        float v = X[(size_t)rrow * 128 + c];
        s += v; s2 += v * v;
    }
    atomicAdd(&sum[c], s);
    atomicAdd(&sq[c], s2);
}

__global__ void bn_apply_kernel(const float* __restrict__ T, const float* __restrict__ sum,
                                const float* __restrict__ sq, const float* __restrict__ g,
                                const float* __restrict__ b, float* out, int do_gelu) {
    long n = (long)NBIG;
    long i = (long)blockIdx.x * blockDim.x + threadIdx.x;
    long stride = (long)gridDim.x * blockDim.x;
    const float invN = 1.0f / (float)N_NODES;
    for (; i < n; i += stride) {
        int c = (int)(i & 127);
        float mean = sum[c] * invN;
        float var  = sq[c] * invN - mean * mean;
        float v = (T[i] - mean) * rsqrtf(var + 1e-5f) * g[c] + b[c];
        if (do_gelu) v = 0.5f * v * (1.0f + erff(v * 0.70710678118654752f));
        out[i] = v;
    }
}

#define FFN_WAVES 2
__global__ void __launch_bounds__(64) ffn_kernel(const float* __restrict__ X,
        const __bf16* W1, const float* bf1, const __bf16* W2, const float* bf2, float* T) {
    __shared__ __bf16 lds[FFN_WAVES * 16 * 256];       // hidden tiles, bf16
    int wave = threadIdx.x >> 5, lane = threadIdx.x & 31;
    int tile = blockIdx.x * FFN_WAVES + wave;   // 1250*2 = 2500 tiles exactly
    int row0 = tile * 16;
    __bf16* hd = lds + wave * 16 * 256;
    int nlo = lane & 15, mhi = (lane >> 4) << 3;

    v16bf a[4];
#pragma unroll
    for (int kt = 0; kt < 4; ++kt) a[kt] = load_a_f32(X, 128, row0, kt * 32, lane);

    // hidden = gelu(X @ W1 + bf1) : 16 x 256 bf16 into per-wave LDS slab
#pragma unroll
    for (int nt = 0; nt < 16; ++nt) {
        v8f acc = {};
#pragma unroll
        for (int kt = 0; kt < 4; ++kt) {
            v16bf b = load_b_bf(W1, 256, kt * 32, nt * 16, lane);
            acc = wmma_bf16(a[kt], b, acc);
        }
        int ncol = nt * 16 + nlo;
        float bs = bf1[ncol];
#pragma unroll
        for (int r = 0; r < 8; ++r) {
            float v = acc[r] + bs;
            v = 0.5f * v * (1.0f + erff(v * 0.70710678118654752f));
            hd[(r + mhi) * 256 + ncol] = bf(v);
        }
    }
    __syncthreads();

    // reload hidden in A layout (K = 256 -> 8 k-tiles) via 16B LDS vector loads
    v16bf a2[8];
    {
        int m = lane & 15, kb8 = (lane >> 4) << 3;
#pragma unroll
        for (int kt = 0; kt < 8; ++kt) {
            const __bf16* p = hd + m * 256 + kt * 32 + kb8;
            v8bf lo = *(const v8bf*)(p);
            v8bf hi = *(const v8bf*)(p + 16);
#pragma unroll
            for (int e = 0; e < 8; ++e) { a2[kt][e] = lo[e]; a2[kt][8 + e] = hi[e]; }
        }
    }

    // T = X + hidden @ W2 + bf2
#pragma unroll
    for (int nt = 0; nt < 8; ++nt) {
        v8f acc = {};
#pragma unroll
        for (int kt = 0; kt < 8; ++kt) {
            v16bf b = load_b_bf(W2, 128, kt * 32, nt * 16, lane);
            acc = wmma_bf16(a2[kt], b, acc);
        }
        int ncol = nt * 16 + nlo;
        float bs = bf2[ncol];
#pragma unroll
        for (int r = 0; r < 8; ++r) {
            size_t idx = (size_t)(row0 + r + mhi) * 128 + ncol;
            T[idx] = acc[r] + bs + X[idx];
        }
    }
}

// ---------------- launcher ----------------

extern "C" void kernel_launch(void* const* d_in, const int* in_sizes, int n_in,
                              void* d_out, int out_size, void* d_ws, size_t ws_size,
                              hipStream_t stream) {
    const float* h   = (const float*)d_in[0];
    const int*   row = (const int*)  d_in[1];
    const int*   col = (const int*)  d_in[2];
    const float* Wq  = (const float*)d_in[3];  const float* bq  = (const float*)d_in[4];
    const float* Wk  = (const float*)d_in[5];  const float* bk  = (const float*)d_in[6];
    const float* Wv  = (const float*)d_in[7];  const float* bv  = (const float*)d_in[8];
    const float* Wo  = (const float*)d_in[9];  const float* bo  = (const float*)d_in[10];
    const float* W1  = (const float*)d_in[11]; const float* bf1 = (const float*)d_in[12];
    const float* W2  = (const float*)d_in[13]; const float* bf2 = (const float*)d_in[14];
    const float* g1  = (const float*)d_in[15]; const float* b1  = (const float*)d_in[16];
    const float* g2  = (const float*)d_in[17]; const float* b2  = (const float*)d_in[18];
    float* out = (float*)d_out;

    // workspace layout (floats); Q reused as AO, K reused as X after attention scores
    float* ws = (float*)d_ws;
    float* Q  = ws;               float* AO = Q;    // attention-output accumulator
    float* K  = ws + NBIG;        float* X  = K;    // post-BN1+GELU activations
    float* V  = ws + 2 * NBIG;
    float* S  = ws + 3 * NBIG;    // E*8 scores -> unnormalized attn (in place)
    float* T  = ws + 4 * NBIG;    // pre-BN buffer (used twice)
    float* Mx = ws + 5 * NBIG;    // N*8 segment max
    float* Z  = Mx + (size_t)N_NODES * 8;   // N*8 segment sum
    float* st = Z  + (size_t)N_NODES * 8;   // 512 floats of BN stats
    // bf16 weight copies (32B-aligned: float offset divisible by 8)
    __bf16* Wqb = (__bf16*)(st + 512);
    __bf16* Wkb = Wqb + 16384;
    __bf16* Wvb = Wkb + 16384;
    __bf16* Wob = Wvb + 16384;
    __bf16* W1b = Wob + 16384;    // 128*256
    __bf16* W2b = W1b + 32768;    // 256*128

    // init reductions + pre-convert weights to bf16 (tiny, L2-resident)
    fill_kernel<<<320, 256, 0, stream>>>(Mx, (long)N_NODES * 8, -1e30f);
    fill_kernel<<<320, 256, 0, stream>>>(Z,  (long)N_NODES * 8, 0.0f);
    fill_kernel<<<2,   256, 0, stream>>>(st, 512, 0.0f);
    cvt_bf16_kernel<<<64, 256, 0, stream>>>(Wq, Wqb, 16384);
    cvt_bf16_kernel<<<64, 256, 0, stream>>>(Wk, Wkb, 16384);
    cvt_bf16_kernel<<<64, 256, 0, stream>>>(Wv, Wvb, 16384);
    cvt_bf16_kernel<<<64, 256, 0, stream>>>(Wo, Wob, 16384);
    cvt_bf16_kernel<<<128, 256, 0, stream>>>(W1, W1b, 32768);
    cvt_bf16_kernel<<<128, 256, 0, stream>>>(W2, W2b, 32768);

    // Q/K/V projections (WMMA bf16)
    qkv_kernel<<<625, 128, 0, stream>>>(h, Wqb, bq, Wkb, bk, Wvb, bv, Q, K, V);

    // per-edge scores + segment max
    edge_score_kernel<<<80000, 256, 0, stream>>>(Q, K, row, col, S, Mx);

    // zero AO (aliases Q; safe: Q dead after edge_score in stream order)
    fill_kernel<<<2048, 256, 0, stream>>>(AO, (long)NBIG, 0.0f);

    // exp + segment sum
    edge_exp_kernel<<<2048, 256, 0, stream>>>(S, Mx, Z, row);

    // normalize + scatter-add SpMM
    edge_spmm_kernel<<<80000, 256, 0, stream>>>(S, Z, V, row, col, AO);

    // output projection + residual -> T (pre-BN1)
    proj_kernel<<<625, 128, 0, stream>>>(AO, Wob, bo, h, T);

    // BN1 stats + apply (with exact GELU) -> X
    bn_stats_kernel<<<256, 128, 0, stream>>>(T, st, st + 128);
    bn_apply_kernel<<<2048, 256, 0, stream>>>(T, st, st + 128, g1, b1, X, 1);

    // fused FFN with residual -> T (pre-BN2)
    ffn_kernel<<<1250, 64, 0, stream>>>(X, W1b, bf1, W2b, bf2, T);

    // BN2 stats + apply -> out
    bn_stats_kernel<<<256, 128, 0, stream>>>(T, st + 256, st + 384);
    bn_apply_kernel<<<2048, 256, 0, stream>>>(T, st + 256, st + 384, g2, b2, out, 0);
}
